// NaiveGRU_5248450036263
// MI455X (gfx1250) — matrix operand order repaired and checked by reading, not compiled, not run on hardware
//
#include <hip/hip_runtime.h>
#include <hip/hip_bf16.h>

// ---------------------------------------------------------------------------
// GRU (SEQ=512, BATCH=64, EMB=512, HID=1024) for gfx1250 (MI455X).
// Persistent WMMA kernel: bf16 GEMMs with f32 accumulate, software grid
// barrier between time steps, fused embedding gather (pre-converted bf16
// table), shared r/z accumulators, next-step embedding prefetch.
// ---------------------------------------------------------------------------

#define VOCAB 32000
#define EMB   512
#define HID   1024
#define OUT_N 5
#define SEQ   512
#define BATCH 64
#define G3    (3 * HID)

typedef __attribute__((ext_vector_type(16))) __bf16 v16bf;
typedef __attribute__((ext_vector_type(8)))  __bf16 v8bf;
typedef __attribute__((ext_vector_type(8)))  float  v8f;

// ---- workspace layout (bytes) ---------------------------------------------
static constexpr size_t OFF_BAR  = 0;                                     // 2 x u32 barrier
static constexpr size_t OFF_WIH  = 256;                                   // bf16 [3H][EMB]
static constexpr size_t OFF_WHH  = OFF_WIH + (size_t)G3 * EMB * 2;        // bf16 [3H][HID]
static constexpr size_t OFF_HF   = OFF_WHH + (size_t)G3 * HID * 2;        // f32  [2][B][HID]
static constexpr size_t OFF_HB   = OFF_HF  + (size_t)2 * BATCH * HID * 4; // bf16 [2][B][HID]
static constexpr size_t OFF_EMBB = OFF_HB  + (size_t)2 * BATCH * HID * 2; // bf16 [VOCAB][EMB]
// total ~43 MB

// ---- helpers ---------------------------------------------------------------
__device__ inline v8f wmma_bf16(v16bf a, v16bf b, v8f c) {
  // v_wmma_f32_16x16x32_bf16  (neg_a, A, neg_b, B, c_mod, C, reuse_a, reuse_b)
  return __builtin_amdgcn_wmma_f32_16x16x32_bf16(false, a, false, b, (short)0, c, false, false);
}

// A-matrix 16x32 bf16 fragment from row-major [M][K]:
// lane row = lane&15; elements 0..7 = K[base+off .. +8), elements 8..15 = K[base+off+16 .. +8)
// with off = 8*(lane>>4).  p points at row*K + base + off.
__device__ inline v16bf load_a_bf16(const __bf16* p) {
  v8bf lo = *(const v8bf*)p;
  v8bf hi = *(const v8bf*)(p + 16);
  return __builtin_shufflevector(lo, hi, 0, 1, 2, 3, 4, 5, 6, 7, 8, 9, 10, 11, 12, 13, 14, 15);
}

__device__ inline float sigmoidf_(float x) { return 1.0f / (1.0f + __expf(-x)); }
__device__ inline float tanhf_(float x)    { return 2.0f / (1.0f + __expf(-2.0f * x)) - 1.0f; }

// Arrive + generation-flip grid barrier (all 256 single-wave blocks resident).
__device__ inline void grid_barrier(unsigned* bar, unsigned nblk) {
  __threadfence();
  if (threadIdx.x == 0) {
    unsigned* cnt = bar;
    unsigned* gen = bar + 1;
    unsigned g = __hip_atomic_load(gen, __ATOMIC_RELAXED, __HIP_MEMORY_SCOPE_AGENT);
    unsigned a = __hip_atomic_fetch_add(cnt, 1u, __ATOMIC_ACQ_REL, __HIP_MEMORY_SCOPE_AGENT);
    if (a == nblk - 1u) {
      __hip_atomic_store(cnt, 0u, __ATOMIC_RELAXED, __HIP_MEMORY_SCOPE_AGENT);
      __hip_atomic_store(gen, g + 1u, __ATOMIC_RELEASE, __HIP_MEMORY_SCOPE_AGENT);
    } else {
      while (__hip_atomic_load(gen, __ATOMIC_ACQUIRE, __HIP_MEMORY_SCOPE_AGENT) == g)
        __builtin_amdgcn_s_sleep(1);
    }
  }
  __syncthreads();
}

// ---- init: barrier reset, f32->bf16 conversions (weights + emb), h0 = 0 ----
__global__ void gru_init_kernel(const float* __restrict__ Wih,
                                const float* __restrict__ Whh,
                                const float* __restrict__ Emb,
                                unsigned char* __restrict__ ws) {
  size_t tid    = (size_t)blockIdx.x * blockDim.x + threadIdx.x;
  size_t stride = (size_t)gridDim.x * blockDim.x;
  if (tid == 0) {
    unsigned* bar = (unsigned*)(ws + OFF_BAR);
    bar[0] = 0u;
    bar[1] = 0u;
  }
  __bf16* wih  = (__bf16*)(ws + OFF_WIH);
  __bf16* whh  = (__bf16*)(ws + OFF_WHH);
  float*  hf   = (float*)(ws + OFF_HF);
  __bf16* hb   = (__bf16*)(ws + OFF_HB);
  __bf16* embb = (__bf16*)(ws + OFF_EMBB);
  for (size_t i = tid; i < (size_t)G3 * EMB; i += stride) wih[i] = (__bf16)Wih[i];
  for (size_t i = tid; i < (size_t)G3 * HID; i += stride) whh[i] = (__bf16)Whh[i];
  for (size_t i = tid; i < (size_t)VOCAB * EMB; i += stride) embb[i] = (__bf16)Emb[i];
  for (size_t i = tid; i < (size_t)2 * BATCH * HID; i += stride) {
    hf[i] = 0.0f;
    hb[i] = (__bf16)0.0f;
  }
}

// ---- persistent GRU kernel --------------------------------------------------
// Grid: 256 blocks x 32 threads. Block b -> tile (m = b>>6 in [0,4), n = b&63 in [0,64)).
// Each wave owns one 16x16 (batch x hidden) tile across all 512 steps.
// Accumulators: acc_r / acc_z shared between x- and h-GEMMs; xn / hn separate.
__global__ void __launch_bounds__(32)
gru_persistent_kernel(const int*   __restrict__ text,
                      const float* __restrict__ b_ih,
                      const float* __restrict__ b_hh,
                      const float* __restrict__ W_fc,
                      const float* __restrict__ b_fc,
                      float*       __restrict__ out,
                      unsigned char* __restrict__ ws) {
  const int lane = threadIdx.x;
  const int m    = blockIdx.x >> 6;  // batch tile
  const int n    = blockIdx.x & 63;  // hidden tile

  unsigned*     bar  = (unsigned*)(ws + OFF_BAR);
  const __bf16* Wih  = (const __bf16*)(ws + OFF_WIH);
  const __bf16* Whh  = (const __bf16*)(ws + OFF_WHH);
  const __bf16* Embb = (const __bf16*)(ws + OFF_EMBB);
  float*  hf0 = (float*)(ws + OFF_HF);
  float*  hf1 = hf0 + (size_t)BATCH * HID;
  __bf16* hb0 = (__bf16*)(ws + OFF_HB);
  __bf16* hb1 = hb0 + (size_t)BATCH * HID;
  float*  hfbuf[2] = {hf0, hf1};
  __bf16* hbbuf[2] = {hb0, hb1};

  const int l16   = lane & 15;
  const int lhi   = lane >> 4;
  const int aoff  = lhi * 8;    // A-matrix K sub-offset
  const int khalf = lhi * 8;    // B-matrix K sub-offset

  const int arow = m * 16 + l16;  // batch row this lane loads for A
  const int ncol = n * 16 + l16;  // per-gate W row / output column this lane touches

  // Combined biases for r/z (x and h accumulators are merged); n kept split.
  const float br   = b_ih[ncol] + b_hh[ncol];
  const float bz   = b_ih[HID + ncol] + b_hh[HID + ncol];
  const float bihn = b_ih[2 * HID + ncol];
  const float bhhn = b_hh[2 * HID + ncol];

  // B-matrix base pointers (row-major W, same 8+8 K split as A per lane per k-step).
  const __bf16* WihR = Wih + (size_t)(0 * HID + ncol) * EMB + khalf;
  const __bf16* WihZ = Wih + (size_t)(1 * HID + ncol) * EMB + khalf;
  const __bf16* WihN = Wih + (size_t)(2 * HID + ncol) * EMB + khalf;
  const __bf16* WhhR = Whh + (size_t)(0 * HID + ncol) * HID + khalf;
  const __bf16* WhhZ = Whh + (size_t)(1 * HID + ncol) * HID + khalf;
  const __bf16* WhhN = Whh + (size_t)(2 * HID + ncol) * HID + khalf;

  int tok = text[0 * BATCH + arow];  // step-0 token for this lane's A row

  for (int s = 0; s < SEQ; ++s) {
    const int cur = s & 1;
    const int nxt = cur ^ 1;

    v8f accr = {0.f, 0.f, 0.f, 0.f, 0.f, 0.f, 0.f, 0.f};
    v8f accz = accr, xn = accr, hn = accr;

    // Prefetch next step's embedding rows (tokens are known ahead of the
    // recurrence) so the gather overlaps this step's WMMA work.
    int tok_next = tok;
    if (s + 1 < SEQ) {
      tok_next = text[(s + 1) * BATCH + arow];
      const __bf16* pf = Embb + (size_t)tok_next * EMB;
      __builtin_prefetch(pf, 0, 1);
      __builtin_prefetch(pf + 256, 0, 1);
    }

    // ---- x side: A = emb_bf16[tok], K = EMB (r/z accumulate into shared acc)
    const __bf16* ap = Embb + (size_t)tok * EMB + aoff;
#pragma unroll 4
    for (int k = 0; k < EMB; k += 32) {
      v16bf a = load_a_bf16(ap + k);
      accr = wmma_bf16(a, load_a_bf16(WihR + k), accr);
      accz = wmma_bf16(a, load_a_bf16(WihZ + k), accz);
      xn   = wmma_bf16(a, load_a_bf16(WihN + k), xn);
    }

    // ---- h side: A = h_prev (bf16), K = HID
    const __bf16* hp = hbbuf[cur] + (size_t)arow * HID + aoff;
#pragma unroll 4
    for (int k = 0; k < HID; k += 32) {
      v16bf a = load_a_bf16(hp + k);
      accr = wmma_bf16(a, load_a_bf16(WhhR + k), accr);
      accz = wmma_bf16(a, load_a_bf16(WhhZ + k), accz);
      hn   = wmma_bf16(a, load_a_bf16(WhhN + k), hn);
    }

    // ---- elementwise GRU update on the D-layout (row = r + 8*lhi, col = l16)
    const float* hcur = hfbuf[cur];
    float*       hnxf = hfbuf[nxt];
    __bf16*      hnxb = hbbuf[nxt];
#pragma unroll
    for (int r = 0; r < 8; ++r) {
      const int brow = m * 16 + r + 8 * lhi;
      const size_t idx = (size_t)brow * HID + ncol;
      float rv = sigmoidf_(accr[r] + br);
      float zv = sigmoidf_(accz[r] + bz);
      float nv = tanhf_((xn[r] + bihn) + rv * (hn[r] + bhhn));
      float hprev = hcur[idx];
      float hnew  = (1.0f - zv) * nv + zv * hprev;
      hnxf[idx] = hnew;
      hnxb[idx] = (__bf16)hnew;
    }

    tok = tok_next;
    grid_barrier(bar, gridDim.x);
  }

  // ---- final FC: out[64][5] = h_last @ W_fc^T + b_fc  (h_last in buffer 0)
  if (blockIdx.x == 0) {
    const float* h = hfbuf[0];
    for (int idx = lane; idx < BATCH * OUT_N; idx += 32) {
      const int b = idx / OUT_N;
      const int o = idx % OUT_N;
      float acc = b_fc[o];
      for (int k = 0; k < HID; ++k) acc += h[(size_t)b * HID + k] * W_fc[(size_t)o * HID + k];
      out[idx] = acc;
    }
  }
}

// ---------------------------------------------------------------------------
extern "C" void kernel_launch(void* const* d_in, const int* in_sizes, int n_in,
                              void* d_out, int out_size, void* d_ws, size_t ws_size,
                              hipStream_t stream) {
  (void)in_sizes; (void)n_in; (void)out_size; (void)ws_size;
  const int*   text = (const int*)d_in[0];
  const float* emb  = (const float*)d_in[1];
  const float* Wih  = (const float*)d_in[2];
  const float* Whh  = (const float*)d_in[3];
  const float* bih  = (const float*)d_in[4];
  const float* bhh  = (const float*)d_in[5];
  const float* Wfc  = (const float*)d_in[6];
  const float* bfc  = (const float*)d_in[7];
  unsigned char* ws = (unsigned char*)d_ws;

  hipLaunchKernelGGL(gru_init_kernel, dim3(2048), dim3(256), 0, stream, Wih, Whh, emb, ws);
  hipLaunchKernelGGL(gru_persistent_kernel, dim3(256), dim3(32), 0, stream,
                     text, bih, bhh, Wfc, bfc, (float*)d_out, ws);
}
